// Attention_49959059587521
// MI455X (gfx1250) — matrix operand order repaired and checked by reading, not compiled, not run on hardware
//
#include <hip/hip_runtime.h>

typedef _Float16 h16v __attribute__((ext_vector_type(16)));
typedef _Float16 h8v  __attribute__((ext_vector_type(8)));
typedef float    f8v  __attribute__((ext_vector_type(8)));
typedef float    f4v  __attribute__((ext_vector_type(4)));

#define Bb 4
#define Nn 2048
#define Cc 512
#define Hh 8
#define HD 64
#define ATT_SCALE 0.125f

__device__ __forceinline__ h16v cat8(h8v lo, h8v hi) {
  h16v r;
#pragma unroll
  for (int i = 0; i < 8; ++i) { r[i] = lo[i]; r[i + 8] = hi[i]; }
  return r;
}

__device__ __forceinline__ f8v wmma_f16(h16v a, h16v b, f8v c) {
  return __builtin_amdgcn_wmma_f32_16x16x32_f16(false, a, false, b, (short)0, c,
                                                false, false);
}

// ---- DPP16 cross-lane (VALU pipe, no LDS traffic) ----
// 0xB1 = quad_perm(1,0,3,2)  -> xor 1
// 0x4E = quad_perm(2,3,0,1)  -> xor 2
// 0x141 = row_half_mirror    -> pairs 4-groups within 8
// 0x140 = row_mirror         -> pairs 8-groups within 16
template <int CTRL>
__device__ __forceinline__ float dpp_mov(float x) {
  int xi = __builtin_bit_cast(int, x);
  int r = __builtin_amdgcn_update_dpp(xi, xi, CTRL, 0xF, 0xF, true);
  return __builtin_bit_cast(float, r);
}
__device__ __forceinline__ float red16_max(float x) {
  x = fmaxf(x, dpp_mov<0xB1>(x));
  x = fmaxf(x, dpp_mov<0x4E>(x));
  x = fmaxf(x, dpp_mov<0x141>(x));
  x = fmaxf(x, dpp_mov<0x140>(x));
  return x;
}
__device__ __forceinline__ float red16_add(float x) {
  x += dpp_mov<0xB1>(x);
  x += dpp_mov<0x4E>(x);
  x += dpp_mov<0x141>(x);
  x += dpp_mov<0x140>(x);
  return x;
}

// ---------------- kernel 1: ksum[b,h,d] = sum_n k2[b,n,h*64+d] ----------------
__global__ __launch_bounds__(256) void ksum_kernel(const float* __restrict__ k2,
                                                   float* __restrict__ ksum) {
  const int bh = blockIdx.x;            // 0..31
  const int b = bh >> 3, h = bh & 7;
  const int d = threadIdx.x & 63;
  const int s = threadIdx.x >> 6;       // 4 slices
  const float* kp = k2 + (size_t)b * Nn * Cc + h * HD + d;
  float acc = 0.f;
  for (int n = s; n < Nn; n += 4) acc += kp[(size_t)n * Cc];
  __shared__ float red[256];
  red[threadIdx.x] = acc;
  __syncthreads();
  if (s == 0)
    ksum[bh * HD + d] = red[d] + red[64 + d] + red[128 + d] + red[192 + d];
}

// ------------- kernel 2: gate[b,h,q] = sigmoid(SCALE * q2 . ksum) -------------
__global__ __launch_bounds__(256) void gate_kernel(const float* __restrict__ q2,
                                                   const float* __restrict__ ksum,
                                                   const int* __restrict__ flag,
                                                   float* __restrict__ gate) {
  const int g = blockIdx.x * 256 + threadIdx.x;   // 0..65535
  const int bh = g >> 11;                         // N=2048 per bh
  const int qi = g & (Nn - 1);
  const int b = bh >> 3, h = bh & 7;
  __shared__ float ks[HD];
  if (threadIdx.x < HD) ks[threadIdx.x] = ksum[bh * HD + threadIdx.x];
  __syncthreads();
  const float* qp = q2 + ((size_t)(b * Nn + qi) * Cc) + h * HD;
  float acc = 0.f;
#pragma unroll 16
  for (int d = 0; d < HD; ++d) acc += qp[d] * ks[d];
  float val = (*flag) ? (1.f / (1.f + __expf(-acc * ATT_SCALE))) : 1.f;
  gate[g] = val;
}

// --------------------- kernel 3: flash attention (WMMA) -----------------------
// Workgroup = 4 waves; wave = 16 query rows; chunk = 64 keys.
// Per chunk: 8 WMMA for S = Q.K^T, online softmax with DPP butterflies,
// 8 WMMA for O += P.V. Dependent WMMA pairs are split into two passes so the
// scheduler can cover the f16-WMMA RAW hazard slots with independent WMMAs.
__global__ __launch_bounds__(128) void attn_fa(const float* __restrict__ Q,
                                               const float* __restrict__ K,
                                               const float* __restrict__ V,
                                               const float* __restrict__ gate,
                                               float* __restrict__ out) {
  __shared__ __align__(16) _Float16 Kt[64][72];      // [key][d]
  __shared__ __align__(16) _Float16 Vt[64][72];      // transposed: [d][key]
  __shared__ __align__(16) _Float16 Pt[4][16][72];   // per-wave [row][key]

  const int wave = threadIdx.x >> 5;
  const int lane = threadIdx.x & 31;
  const int n    = lane & 15;
  const int hig  = lane >> 4;

  const int bh   = blockIdx.x >> 5;    // B*H = 32
  const int qblk = blockIdx.x & 31;    // 32 query blocks of 64
  const int b = bh >> 3, h = bh & 7;
  const int qtile = qblk * 64 + wave * 16;

  // ---- Q A-fragments (row = n), SCALE folded in, f32 -> f16 ----
  const float* qrow = Q + ((size_t)(b * Nn + qtile + n) * Cc) + h * HD;
  h16v qa0, qa1;
  {
    const int d0 = hig * 8;
#pragma unroll
    for (int f = 0; f < 2; ++f) {
      const int base = f * 32 + d0;
      f4v t0 = *(const f4v*)(qrow + base);
      f4v t1 = *(const f4v*)(qrow + base + 4);
      f4v t2 = *(const f4v*)(qrow + base + 16);
      f4v t3 = *(const f4v*)(qrow + base + 20);
      h16v a;
#pragma unroll
      for (int i = 0; i < 4; ++i) {
        a[i]      = (_Float16)(t0[i] * ATT_SCALE);
        a[4 + i]  = (_Float16)(t1[i] * ATT_SCALE);
        a[8 + i]  = (_Float16)(t2[i] * ATT_SCALE);
        a[12 + i] = (_Float16)(t3[i] * ATT_SCALE);
      }
      if (f == 0) qa0 = a; else qa1 = a;
    }
  }

  f8v O0 = {}, O1 = {}, O2 = {}, O3 = {};
  float m_run[8], l_run[8];
#pragma unroll
  for (int r = 0; r < 8; ++r) { m_run[r] = -1e30f; l_run[r] = 0.f; }

  for (int kc = 0; kc < Nn; kc += 64) {
    __syncthreads();
    // ---- cooperative K/V tile load: 64 keys x 64 d, f32 -> f16 ----
#pragma unroll
    for (int i = 0; i < 2; ++i) {
      const int key = (threadIdx.x >> 2) + i * 32;
      const int seg = (threadIdx.x & 3) * 16;
      const size_t rbase = ((size_t)(b * Nn + kc + key) * Cc) + h * HD + seg;
      const float* kp = K + rbase;
      f4v c0 = *(const f4v*)(kp);
      f4v c1 = *(const f4v*)(kp + 4);
      f4v c2 = *(const f4v*)(kp + 8);
      f4v c3 = *(const f4v*)(kp + 12);
      h8v p0, p1;
#pragma unroll
      for (int j = 0; j < 4; ++j) {
        p0[j]     = (_Float16)c0[j];
        p0[4 + j] = (_Float16)c1[j];
        p1[j]     = (_Float16)c2[j];
        p1[4 + j] = (_Float16)c3[j];
      }
      *(h8v*)&Kt[key][seg]     = p0;
      *(h8v*)&Kt[key][seg + 8] = p1;

      const float* vp = V + rbase;
      f4v d0 = *(const f4v*)(vp);
      f4v d1 = *(const f4v*)(vp + 4);
      f4v d2 = *(const f4v*)(vp + 8);
      f4v d3 = *(const f4v*)(vp + 12);
      float vv[16];
#pragma unroll
      for (int j = 0; j < 4; ++j) {
        vv[j] = d0[j]; vv[4 + j] = d1[j]; vv[8 + j] = d2[j]; vv[12 + j] = d3[j];
      }
#pragma unroll
      for (int j = 0; j < 16; ++j)
        Vt[seg + j][key] = (_Float16)vv[j];   // transposed store

      if (kc + 64 < Nn) {   // stream next chunk toward L2/WGP$
        __builtin_prefetch(kp + (size_t)64 * Cc, 0, 0);
        __builtin_prefetch(vp + (size_t)64 * Cc, 0, 0);
      }
    }
    __syncthreads();

    // ---- scores: pass 1 (K-step 0) for all 4 key-groups, then pass 2 ----
    f8v S0, S1, S2, S3;
#pragma unroll
    for (int g = 0; g < 4; ++g) {
      h16v b0 = cat8(*(const h8v*)&Kt[g * 16 + n][hig * 8],
                     *(const h8v*)&Kt[g * 16 + n][16 + hig * 8]);
      f8v s = {};
      s = wmma_f16(qa0, b0, s);
      if (g == 0) S0 = s; else if (g == 1) S1 = s;
      else if (g == 2) S2 = s; else S3 = s;
    }
#pragma unroll
    for (int g = 0; g < 4; ++g) {
      h16v b1 = cat8(*(const h8v*)&Kt[g * 16 + n][32 + hig * 8],
                     *(const h8v*)&Kt[g * 16 + n][48 + hig * 8]);
      f8v s = (g == 0) ? S0 : (g == 1) ? S1 : (g == 2) ? S2 : S3;
      s = wmma_f16(qa1, b1, s);
      if (g == 0) S0 = s; else if (g == 1) S1 = s;
      else if (g == 2) S2 = s; else S3 = s;
    }

    // ---- online softmax: DPP 16-lane butterflies (VALU pipe) ----
    float corr[8];
#pragma unroll
    for (int r = 0; r < 8; ++r) {
      float mx = fmaxf(fmaxf(S0[r], S1[r]), fmaxf(S2[r], S3[r]));
      mx = red16_max(mx);
      float mnew = fmaxf(m_run[r], mx);
      float c  = __expf(m_run[r] - mnew);
      float p0 = __expf(S0[r] - mnew);
      float p1 = __expf(S1[r] - mnew);
      float p2 = __expf(S2[r] - mnew);
      float p3 = __expf(S3[r] - mnew);
      S0[r] = p0; S1[r] = p1; S2[r] = p2; S3[r] = p3;
      float rs = red16_add((p0 + p1) + (p2 + p3));
      l_run[r] = l_run[r] * c + rs;
      m_run[r] = mnew;
      corr[r] = c;
    }
#pragma unroll
    for (int r = 0; r < 8; ++r) {
      O0[r] *= corr[r]; O1[r] *= corr[r]; O2[r] *= corr[r]; O3[r] *= corr[r];
    }

    // ---- P: C-layout -> A-layout transpose via per-wave LDS tile ----
#pragma unroll
    for (int r = 0; r < 8; ++r) {
      const int m = r + hig * 8;
      Pt[wave][m][n]      = (_Float16)S0[r];
      Pt[wave][m][16 + n] = (_Float16)S1[r];
      Pt[wave][m][32 + n] = (_Float16)S2[r];
      Pt[wave][m][48 + n] = (_Float16)S3[r];
    }
    // same-wave DS ops are in-order (ISA 7.3): safe to read back
    h16v pa0 = cat8(*(const h8v*)&Pt[wave][n][hig * 8],
                    *(const h8v*)&Pt[wave][n][16 + hig * 8]);
    h16v pa1 = cat8(*(const h8v*)&Pt[wave][n][32 + hig * 8],
                    *(const h8v*)&Pt[wave][n][48 + hig * 8]);

    // ---- O += P . V : pass 1 (keys 0..31) all d-groups, then pass 2 ----
#pragma unroll
    for (int dg = 0; dg < 4; ++dg) {
      const int d = dg * 16 + n;
      h16v vb0 = cat8(*(const h8v*)&Vt[d][hig * 8],
                      *(const h8v*)&Vt[d][16 + hig * 8]);
      f8v acc = (dg == 0) ? O0 : (dg == 1) ? O1 : (dg == 2) ? O2 : O3;
      acc = wmma_f16(pa0, vb0, acc);
      if (dg == 0) O0 = acc; else if (dg == 1) O1 = acc;
      else if (dg == 2) O2 = acc; else O3 = acc;
    }
#pragma unroll
    for (int dg = 0; dg < 4; ++dg) {
      const int d = dg * 16 + n;
      h16v vb1 = cat8(*(const h8v*)&Vt[d][32 + hig * 8],
                      *(const h8v*)&Vt[d][48 + hig * 8]);
      f8v acc = (dg == 0) ? O0 : (dg == 1) ? O1 : (dg == 2) ? O2 : O3;
      acc = wmma_f16(pa1, vb1, acc);
      if (dg == 0) O0 = acc; else if (dg == 1) O1 = acc;
      else if (dg == 2) O2 = acc; else O3 = acc;
    }
  }

  // ---- finalize: normalize by l, apply gate, store fp32 ----
  const float* grow = gate + (size_t)bh * Nn + qtile;
#pragma unroll
  for (int r = 0; r < 8; ++r) {
    const int m = r + hig * 8;
    const float sc = grow[m] / l_run[r];
    float* orow = out + ((size_t)(b * Nn + qtile + m) * Cc) + h * HD;
    orow[n]      = O0[r] * sc;
    orow[16 + n] = O1[r] * sc;
    orow[32 + n] = O2[r] * sc;
    orow[48 + n] = O3[r] * sc;
  }
}

extern "C" void kernel_launch(void* const* d_in, const int* in_sizes, int n_in,
                              void* d_out, int out_size, void* d_ws,
                              size_t ws_size, hipStream_t stream) {
  (void)in_sizes; (void)n_in; (void)out_size; (void)ws_size;
  const float* q    = (const float*)d_in[0];
  const float* k    = (const float*)d_in[1];
  const float* v    = (const float*)d_in[2];
  const float* q2   = (const float*)d_in[3];
  const float* k2   = (const float*)d_in[4];
  const int*   flag = (const int*)d_in[5];
  float* out = (float*)d_out;

  float* ksum = (float*)d_ws;                 // B*H*HD   = 2048 floats
  float* gate = ksum + Bb * Hh * HD;          // B*H*N    = 65536 floats

  ksum_kernel<<<Bb * Hh, 256, 0, stream>>>(k2, ksum);
  gate_kernel<<<(Bb * Hh * Nn) / 256, 256, 0, stream>>>(q2, ksum, flag, gate);
  attn_fa<<<Bb * Hh * (Nn / 64), 128, 0, stream>>>(q, k, v, gate, out);
}
